// Attention_32435593019406
// MI455X (gfx1250) — matrix operand order repaired and checked by reading, not compiled, not run on hardware
//
#include <hip/hip_runtime.h>

typedef __bf16 bf16;
typedef __attribute__((ext_vector_type(16))) __bf16 v16bf;
typedef __attribute__((ext_vector_type(8)))  float  v8f;
typedef __attribute__((ext_vector_type(4)))  int    v4i;

#define B_   2
#define N_   2048
#define DIM_ 512
#define H_   8
#define DH_  64
#define SCALE_ 0.125f
#define NEGMAX_ (-3.4028235e38f)

#if __has_builtin(__builtin_amdgcn_global_load_async_to_lds_b128)
#define USE_ASYNC 1
#else
#define USE_ASYNC 0
#endif

// ---------------------------------------------------------------- helpers ---

__device__ __forceinline__ v8f splat8(float x) {
    v8f r;
#pragma unroll
    for (int i = 0; i < 8; ++i) r[i] = x;
    return r;
}

__device__ __forceinline__ v8f wmma_bf16(v16bf a, v16bf b, v8f c) {
    return __builtin_amdgcn_wmma_f32_16x16x32_bf16(
        /*neg_a=*/false, a, /*neg_b=*/false, b,
        /*c_mod=*/(short)0, c, /*reuse_a=*/false, /*reuse_b=*/false);
}

// 16-byte memory -> LDS copy, one lane, async (ASYNCcnt) when available.
__device__ __forceinline__ void cp16(const bf16* g, bf16* l) {
#if USE_ASYNC
    __builtin_amdgcn_global_load_async_to_lds_b128(
        (__attribute__((address_space(1))) v4i*)g,
        (__attribute__((address_space(3))) v4i*)l, 0, 0);
#else
    *(uint4*)l = *(const uint4*)g;
#endif
}

__device__ __forceinline__ void wait_async0() {
#if USE_ASYNC
#if __has_builtin(__builtin_amdgcn_s_wait_asynccnt)
    __builtin_amdgcn_s_wait_asynccnt(0);
#else
    asm volatile("s_wait_asynccnt 0" ::: "memory");
#endif
#endif
}

// A fragment (16x32 bf16, M x K). Tile in LDS is row-major [rows][ld] with K
// contiguous. Lane layout (ISA 7.12.2): lane m=l%16 holds row m; half h=l/16
// holds K in [8h,8h+8) and [16+8h,16+8h+8); VGPR v packs K pair (2 bf16).
__device__ __forceinline__ v16bf frag_a(const bf16* t, int ld, int lane) {
    const int m = lane & 15, h = (lane >> 4) & 1;
    const bf16* r = t + m * ld;
    v16bf a;
#pragma unroll
    for (int v = 0; v < 8; ++v) {
        const int kk = (v < 4) ? (8 * h + 2 * v) : (16 + 8 * h + 2 * (v - 4));
        a[2 * v]     = r[kk];
        a[2 * v + 1] = r[kk + 1];
    }
    return a;
}

// B fragment (32x16 bf16, K x N). Tile stored "N-major, K-contiguous":
// t[n*ld + k] == B[k][n]. Lane n=l%16 holds column n; half h holds K in
// [8h,8h+8) and [16+8h,16+8h+8); VGPR v packs K pair (2 bf16).
__device__ __forceinline__ v16bf frag_b(const bf16* t, int ld, int lane) {
    const int n = lane & 15, h = (lane >> 4) & 1;
    const bf16* r = t + n * ld;
    v16bf b;
#pragma unroll
    for (int v = 0; v < 8; ++v) {
        const int kk = (v < 4) ? (8 * h + 2 * v) : (16 + 8 * h + 2 * (v - 4));
        b[2 * v]     = r[kk];
        b[2 * v + 1] = r[kk + 1];
    }
    return b;
}

// ------------------------------------------------------------ convert f32->bf16

__global__ __launch_bounds__(256) void cvt_kernel(const float* __restrict__ s,
                                                  bf16* __restrict__ d, int n) {
    int i = blockIdx.x * 256 + threadIdx.x;
    if (i < n) d[i] = (bf16)s[i];
}

// ------------------------------------------------------------------- GEMM ----
// C[M,Ncols] = A[M,K] @ Bw[K,Ncols] + bias.  128 threads (4 waves), WG tile
// 128x64, wave tile 32x64, K-step 32.  mode 0: scatter bf16 into per-head
// Q/K/V buffers (virtual column = colbase+gc in [wq|wkv] space).  mode 1:
// write f32 to Of (final projection).

__global__ __launch_bounds__(128) void gemm_kernel(
    const bf16* __restrict__ A, const bf16* __restrict__ Bw,
    const float* __restrict__ bias, int M, int K, int Ncols, int colbase,
    int mode, bf16* __restrict__ Qb, bf16* __restrict__ Kb,
    bf16* __restrict__ Vb, float* __restrict__ Of) {
    __shared__ bf16 As[128 * 32];
    __shared__ bf16 Bt[64 * 32];  // transposed: Bt[n][k]

    const int m0 = blockIdx.x * 128, n0 = blockIdx.y * 64;
    const int tid = threadIdx.x, lane = tid & 31, w = tid >> 5;
    const int nidx = lane & 15, h2 = (lane >> 4) & 1;

    v8f acc[2][4];
#pragma unroll
    for (int i = 0; i < 2; ++i)
#pragma unroll
        for (int ct = 0; ct < 4; ++ct)
            acc[i][ct] = splat8(bias[n0 + 16 * ct + nidx]);

    for (int ks = 0; ks < K / 32; ++ks) {
        __syncthreads();
        // A tile 128x32: 16-byte chunks (async to LDS when available)
#pragma unroll
        for (int c = tid; c < 512; c += 128) {
            const int r = c >> 2, c8 = (c & 3) << 3;
            cp16(A + (unsigned)(m0 + r) * K + (unsigned)(ks * 32 + c8),
                 As + r * 32 + c8);
        }
        // B tile 32x64 -> transposed Bt[n][k]
#pragma unroll 2
        for (int c = tid; c < 512; c += 128) {
            const int k = c >> 4, c4 = (c & 15) << 2;
            union { uint2 u; bf16 b[4]; } cv;
            cv.u = *(const uint2*)(Bw + (unsigned)(ks * 32 + k) * Ncols +
                                   (unsigned)(n0 + c4));
#pragma unroll
            for (int i = 0; i < 4; ++i) Bt[(c4 + i) * 32 + k] = cv.b[i];
        }
        wait_async0();
        __syncthreads();

        const v16bf af0 = frag_a(As + (32 * w) * 32, 32, lane);
        const v16bf af1 = frag_a(As + (32 * w + 16) * 32, 32, lane);
#pragma unroll
        for (int ct = 0; ct < 4; ++ct) {
            const v16bf bf = frag_b(Bt + (16 * ct) * 32, 32, lane);
            acc[0][ct] = wmma_bf16(af0, bf, acc[0][ct]);
            acc[1][ct] = wmma_bf16(af1, bf, acc[1][ct]);
        }
    }

    // Epilogue. D layout: lane l, VGPR v -> row v + 8*(l/16), col l%16.
#pragma unroll
    for (int i = 0; i < 2; ++i)
#pragma unroll
        for (int ct = 0; ct < 4; ++ct)
#pragma unroll
            for (int v = 0; v < 8; ++v) {
                const int gr = m0 + 32 * w + 16 * i + v + 8 * h2;
                const int gc = colbase + n0 + 16 * ct + nidx;
                const float val = acc[i][ct][v];
                if (mode == 0) {
                    const int bi = gr >> 11, n = gr & (N_ - 1);
                    int c = gc;
                    bf16* dst;
                    if (c < 512)       { dst = Qb; }
                    else if (c < 1024) { dst = Kb; c -= 512; }
                    else               { dst = Vb; c -= 1024; }
                    const int hh = c >> 6, dd = c & 63;
                    dst[(((unsigned)bi * H_ + hh) * N_ + n) * DH_ + dd] = (bf16)val;
                } else {
                    Of[(unsigned)gr * DIM_ + (unsigned)gc] = val;
                }
            }
}

// -------------------------------------------------------- flash attention ----
// grid: (N/64, B*H). 128 threads = 4 waves; wave owns 16 query rows; WG owns
// 64. Online softmax over 32 key-blocks of 64.

__global__ __launch_bounds__(128) void attn_kernel(
    const bf16* __restrict__ Qb, const bf16* __restrict__ Kb,
    const bf16* __restrict__ Vb, const unsigned char* __restrict__ mask,
    bf16* __restrict__ Ob) {
    __shared__ bf16 Qs[64 * 64];
    __shared__ bf16 Ks[64 * 64];       // [key][dh]
    __shared__ bf16 Vt[64 * 64];       // transposed: [dh][key]
    __shared__ bf16 Ps[4 * 16 * 64];   // per-wave P scratch [16][64]

    const int qblk = blockIdx.x;            // 0..31
    const int bh = blockIdx.y;              // 0..15 == (b,h)
    const int b = bh >> 3;
    const int tid = threadIdx.x, lane = tid & 31, w = tid >> 5;
    const int nidx = lane & 15, h2 = (lane >> 4) & 1;

    const bf16* Qg = Qb + (unsigned)bh * (N_ * DH_) + (unsigned)qblk * (64 * DH_);
    const bf16* Kg = Kb + (unsigned)bh * (N_ * DH_);
    const bf16* Vg = Vb + (unsigned)bh * (N_ * DH_);
    const unsigned char* mb = mask + (unsigned)b * (unsigned)(N_ * N_);

    // stage Q tile (64x64) once
#pragma unroll
    for (int c = tid; c < 512; c += 128) cp16(Qg + (c << 3), Qs + (c << 3));
    wait_async0();
    __syncthreads();

    v16bf qf[2];
    qf[0] = frag_a(Qs + (16 * w) * 64 + 0, 64, lane);
    qf[1] = frag_a(Qs + (16 * w) * 64 + 32, 64, lane);

    float m_i[8], l_i[8];
    v8f oacc[4];
#pragma unroll
    for (int v = 0; v < 8; ++v) { m_i[v] = NEGMAX_; l_i[v] = 0.f; }
#pragma unroll
    for (int dt = 0; dt < 4; ++dt) oacc[dt] = splat8(0.f);

    const int rowbase = qblk * 64 + 16 * w + 8 * h2;  // + v = query index i

    for (int kb = 0; kb < N_ / 64; ++kb) {
        __syncthreads();
        // K tile [key][dh]: async 16B chunks
#pragma unroll
        for (int c = tid; c < 512; c += 128)
            cp16(Kg + (unsigned)kb * (64 * DH_) + (c << 3), Ks + (c << 3));
        // V tile transposed -> Vt[dh][key] (VALU work overlaps async K copy)
        {
            const bf16* src = Vg + (unsigned)kb * (64 * DH_);
#pragma unroll 2
            for (int c = tid; c < 1024; c += 128) {
                const int k = c >> 4, d4 = (c & 15) << 2;
                union { uint2 u; bf16 b[4]; } cv;
                cv.u = *(const uint2*)(src + k * DH_ + d4);
#pragma unroll
                for (int i = 0; i < 4; ++i) Vt[(d4 + i) * 64 + k] = cv.b[i];
            }
        }
        if (kb + 1 < N_ / 64) {  // global_prefetch_b8 next block
            __builtin_prefetch(Kg + (unsigned)(kb + 1) * (64 * DH_) + tid * 32, 0, 0);
            __builtin_prefetch(Vg + (unsigned)(kb + 1) * (64 * DH_) + tid * 32, 0, 0);
        }
        wait_async0();
        __syncthreads();

        // sim = Q K^T : 4 column tiles x 2 k-steps
        v8f st[4];
#pragma unroll
        for (int ct = 0; ct < 4; ++ct) {
            v8f a = splat8(0.f);
            a = wmma_bf16(qf[0], frag_b(Ks + (16 * ct) * 64 + 0, 64, lane), a);
            a = wmma_bf16(qf[1], frag_b(Ks + (16 * ct) * 64 + 32, 64, lane), a);
            st[ct] = a;
        }

        // scale + mask (32-bit offsets, base pre-biased by batch)
#pragma unroll
        for (int ct = 0; ct < 4; ++ct) {
            const unsigned moff0 =
                (unsigned)rowbase * N_ + (unsigned)(kb * 64 + 16 * ct + nidx);
#pragma unroll
            for (int v = 0; v < 8; ++v) {
                const float s = st[ct][v] * SCALE_;
                st[ct][v] = mb[moff0 + (unsigned)v * N_] ? s : NEGMAX_;
            }
        }

        // row max (rows of a D-tile live in one 16-lane half)
        float mt[8];
#pragma unroll
        for (int v = 0; v < 8; ++v) {
            float m = fmaxf(fmaxf(st[0][v], st[1][v]), fmaxf(st[2][v], st[3][v]));
#pragma unroll
            for (int off = 8; off >= 1; off >>= 1)
                m = fmaxf(m, __shfl_xor(m, off, 32));
            mt[v] = m;
        }

        float alpha[8];
#pragma unroll
        for (int v = 0; v < 8; ++v) {
            const float mn = fmaxf(m_i[v], mt[v]);
            alpha[v] = __expf(m_i[v] - mn);
            m_i[v] = mn;
        }

        // exp + row sums
        float sum[8];
#pragma unroll
        for (int v = 0; v < 8; ++v) sum[v] = 0.f;
#pragma unroll
        for (int ct = 0; ct < 4; ++ct)
#pragma unroll
            for (int v = 0; v < 8; ++v) {
                const float p = __expf(st[ct][v] - m_i[v]);
                st[ct][v] = p;
                sum[v] += p;
            }
#pragma unroll
        for (int v = 0; v < 8; ++v) {
            float s = sum[v];
#pragma unroll
            for (int off = 8; off >= 1; off >>= 1) s += __shfl_xor(s, off, 32);
            l_i[v] = l_i[v] * alpha[v] + s;
        }
#pragma unroll
        for (int dt = 0; dt < 4; ++dt)
#pragma unroll
            for (int v = 0; v < 8; ++v) oacc[dt][v] *= alpha[v];

        // P: D-layout -> LDS -> A-layout (per-wave region, wave-internal sync)
        bf16* Pw = Ps + w * 1024;
#pragma unroll
        for (int ct = 0; ct < 4; ++ct)
#pragma unroll
            for (int v = 0; v < 8; ++v)
                Pw[(v + 8 * h2) * 64 + 16 * ct + nidx] = (bf16)st[ct][v];
        asm volatile("s_wait_dscnt 0" ::: "memory");

        const v16bf pf0 = frag_a(Pw + 0, 64, lane);
        const v16bf pf1 = frag_a(Pw + 32, 64, lane);

        // O += P @ V : 4 dh tiles x 2 k-steps (Vt is [dh][key] -> frag_b direct)
#pragma unroll
        for (int dt = 0; dt < 4; ++dt) {
            oacc[dt] = wmma_bf16(pf0, frag_b(Vt + (16 * dt) * 64 + 0, 64, lane), oacc[dt]);
            oacc[dt] = wmma_bf16(pf1, frag_b(Vt + (16 * dt) * 64 + 32, 64, lane), oacc[dt]);
        }
    }

    // normalize + store O as [b][n][h][dh] bf16 (row = contiguous INNER)
    const int head = bh & 7;
#pragma unroll
    for (int dt = 0; dt < 4; ++dt)
#pragma unroll
        for (int v = 0; v < 8; ++v) {
            const int i = rowbase + v;
            const int d = 16 * dt + nidx;
            Ob[(((unsigned)b * N_ + (unsigned)i) * H_ + head) * DH_ + d] =
                (bf16)(oacc[dt][v] / l_i[v]);
        }
}

// ---------------------------------------------------------------- launch ----

extern "C" void kernel_launch(void* const* d_in, const int* in_sizes, int n_in,
                              void* d_out, int out_size, void* d_ws, size_t ws_size,
                              hipStream_t stream) {
    const float* nodes = (const float*)d_in[0];
    const unsigned char* emask = (const unsigned char*)d_in[1];  // jnp bool_
    const float* wq  = (const float*)d_in[2];
    const float* bq  = (const float*)d_in[3];
    const float* wkv = (const float*)d_in[4];
    const float* bkv = (const float*)d_in[5];
    const float* wo  = (const float*)d_in[6];
    const float* bo  = (const float*)d_in[7];
    float* out = (float*)d_out;

    char* p = (char*)d_ws;
    bf16* Ab   = (bf16*)p; p += (size_t)4096 * 512 * 2;   // nodes bf16
    bf16* Wqb  = (bf16*)p; p += (size_t)512 * 512 * 2;
    bf16* Wkvb = (bf16*)p; p += (size_t)512 * 1024 * 2;
    bf16* Wob  = (bf16*)p; p += (size_t)512 * 512 * 2;
    bf16* Qb   = (bf16*)p; p += (size_t)B_ * H_ * N_ * DH_ * 2;
    bf16* Kb   = (bf16*)p; p += (size_t)B_ * H_ * N_ * DH_ * 2;
    bf16* Vb   = (bf16*)p; p += (size_t)B_ * H_ * N_ * DH_ * 2;
    bf16* Ob   = (bf16*)p; p += (size_t)4096 * 512 * 2;

    cvt_kernel<<<(4096 * 512 + 255) / 256, 256, 0, stream>>>(nodes, Ab, 4096 * 512);
    cvt_kernel<<<(512 * 512 + 255) / 256, 256, 0, stream>>>(wq, Wqb, 512 * 512);
    cvt_kernel<<<(512 * 1024 + 255) / 256, 256, 0, stream>>>(wkv, Wkvb, 512 * 1024);
    cvt_kernel<<<(512 * 512 + 255) / 256, 256, 0, stream>>>(wo, Wob, 512 * 512);

    // Q = nodes @ wq + bq            (virtual cols [0,512) -> Q)
    gemm_kernel<<<dim3(32, 8), 128, 0, stream>>>(
        Ab, Wqb, bq, 4096, 512, 512, /*colbase=*/0, /*mode=*/0,
        Qb, Kb, Vb, nullptr);
    // K,V = nodes @ wkv + bkv        (virtual cols [512,1536) -> K,V)
    gemm_kernel<<<dim3(32, 16), 128, 0, stream>>>(
        Ab, Wkvb, bkv, 4096, 512, 1024, /*colbase=*/512, /*mode=*/0,
        Qb, Kb, Vb, nullptr);

    attn_kernel<<<dim3(N_ / 64, B_ * H_), 128, 0, stream>>>(Qb, Kb, Vb, emask, Ob);

    // out = O @ wo + bo (f32)
    gemm_kernel<<<dim3(32, 8), 128, 0, stream>>>(
        Ob, Wob, bo, 4096, 512, 512, /*colbase=*/0, /*mode=*/1,
        nullptr, nullptr, nullptr, out);
}
